// AttentionPooling_58016418234866
// MI455X (gfx1250) — matrix operand order repaired and checked by reading, not compiled, not run on hardware
//
#include <hip/hip_runtime.h>
#include <hip/hip_bf16.h>

// ---------------------------------------------------------------------------
// AttentionPooling on MI455X (gfx1250, wave32)
//  Pass A: scores = tanh(X @ W1 + b1) @ W2 + b2   via v_wmma_f32_16x16x32_bf16
//  Pass B/C: global max of scores
//  Pass D: Z = sum exp(s-M); seg[b] = sum_{i in b} exp(s_i - M)   (Z cancels)
//  Pass D2: w_norm_i = exp(s_i - M) / (seg[b_i] + EPS*Z)
//  Pass E: out[b,:] += w_norm_i * x[i,:]  (sorted-batch register accumulation)
// ---------------------------------------------------------------------------

typedef __attribute__((ext_vector_type(16))) __bf16 v16bf;
typedef __attribute__((ext_vector_type(8)))  __bf16 v8bf;
typedef __attribute__((ext_vector_type(8)))  float  v8f;
typedef __attribute__((ext_vector_type(4)))  float  f32x4;

#define HDIM   256
#define HHALF  128
#define LDS_STRIDE 264   // 256 + 8 halves pad -> 4-bank skew per row, conflict-free b128
#define CHUNK  1024

// ------------------------- Pass A: scores via WMMA --------------------------
__global__ void __launch_bounds__(256)
scores_kernel(const float* __restrict__ X,
              const float* __restrict__ W1,
              const float* __restrict__ b1,
              const float* __restrict__ W2,
              const float* __restrict__ b2,
              float* __restrict__ scores,
              int N, int numTiles)
{
    extern __shared__ __bf16 w1t[];            // [128 cols][264 halves] = W1^T bf16
    const int tid = threadIdx.x;

    // Stage W1^T (column-major W1) into LDS as bf16, coalesced global reads.
    for (int idx = tid; idx < HDIM * HHALF; idx += 256) {
        int k = idx >> 7;          // 0..255
        int c = idx & 127;         // 0..127
        w1t[c * LDS_STRIDE + k] = (__bf16)W1[k * HHALF + c];
    }

    const int wave = tid >> 5;     // 0..7  (one 16-row tile per wave)
    const int lane = tid & 31;
    const int l16  = lane & 15;
    const int half = lane >> 4;    // 0/1
    const int koff = half * 8;

    // Per-lane epilogue constants: b1[h], W2[h] for h = t*16 + l16
    float b1v[8], w2v[8];
    #pragma unroll
    for (int t = 0; t < 8; ++t) {
        int h = t * 16 + l16;
        b1v[t] = b1[h];
        w2v[t] = W2[h];
    }
    const float bias2 = b2[0];

    for (int g = blockIdx.x; g < numTiles; g += gridDim.x) {
        // Barrier each iteration: (a) covers the initial W1 staging, and
        // (b) acts as an LDS fence so LICM cannot hoist the 64 B-fragment
        // ds_loads out of this loop (hoisting spilled 512 VGPRs to scratch,
        // turning every WMMA operand fetch into a scratch_load_b128).
        __syncthreads();

        const long rowBase = (long)g * 128 + (long)wave * 16;
        long row  = rowBase + l16;
        long rowc = (row < N) ? row : (long)(N - 1);     // clamp tail (store predicated)
        const float* xrow = X + rowc * HDIM;

        v8f acc[8];
        #pragma unroll
        for (int t = 0; t < 8; ++t) acc[t] = (v8f)0.0f;

        #pragma unroll
        for (int k = 0; k < 8; ++k) {
            const int kb = k * 32;
            // A fragment (16x32 bf16): lanes 0-15 hold K kb+0..7 & kb+16..23,
            // lanes 16-31 hold K kb+8..15 & kb+24..31 (row = rowBase + l16).
            f32x4 f0 = *(const f32x4*)(xrow + kb + koff);
            f32x4 f1 = *(const f32x4*)(xrow + kb + koff + 4);
            f32x4 f2 = *(const f32x4*)(xrow + kb + 16 + koff);
            f32x4 f3 = *(const f32x4*)(xrow + kb + 16 + koff + 4);
            v16bf a;
            #pragma unroll
            for (int j = 0; j < 4; ++j) {
                a[j]      = (__bf16)f0[j];
                a[4 + j]  = (__bf16)f1[j];
                a[8 + j]  = (__bf16)f2[j];
                a[12 + j] = (__bf16)f3[j];
            }

            // B fragments (32x16 bf16): col n = t*16 + l16,
            // lanes 0-15 K kb+0..7 & kb+16..23, lanes 16-31 K kb+8..15 & kb+24..31.
            // Double-buffered over t so the t+1 ds_loads are in flight while
            // WMMA t executes (wait becomes dscnt<=2 instead of dscnt==0).
            const __bf16* bbase = &w1t[l16 * LDS_STRIDE + kb + half * 8];
            v8bf blo = *(const v8bf*)(bbase);
            v8bf bhi = *(const v8bf*)(bbase + 16);
            #pragma unroll
            for (int t = 0; t < 8; ++t) {
                v8bf blon = blo, bhin = bhi;
                if (t < 7) {               // compile-time under full unroll
                    const __bf16* bp = bbase + (t + 1) * 16 * LDS_STRIDE;
                    blon = *(const v8bf*)bp;
                    bhin = *(const v8bf*)(bp + 16);
                }
                v16bf bb;
                #pragma unroll
                for (int j = 0; j < 8; ++j) { bb[j] = blo[j]; bb[8 + j] = bhi[j]; }
                acc[t] = __builtin_amdgcn_wmma_f32_16x16x32_bf16(
                    false, a, false, bb, (short)0, acc[t], false, false);
                blo = blon; bhi = bhin;
            }
        }

        // Epilogue: C layout -> element (M = v + 8*half, h = t*16 + l16).
        float partial[8];
        #pragma unroll
        for (int v = 0; v < 8; ++v) partial[v] = 0.0f;
        #pragma unroll
        for (int t = 0; t < 8; ++t) {
            #pragma unroll
            for (int v = 0; v < 8; ++v) {
                float hv = tanhf(acc[t][v] + b1v[t]);
                partial[v] += hv * w2v[t];
            }
        }
        #pragma unroll
        for (int v = 0; v < 8; ++v) {
            float r = partial[v];
            r += __shfl_xor(r, 1, 16);
            r += __shfl_xor(r, 2, 16);
            r += __shfl_xor(r, 4, 16);
            r += __shfl_xor(r, 8, 16);
            long orow = rowBase + v + half * 8;
            if (l16 == 0 && orow < N) scores[orow] = r + bias2;
        }
    }
}

// ---------------- Pass B: per-block max (+ zero output buffer) --------------
__global__ void __launch_bounds__(256)
reduce_max_kernel(const float* __restrict__ scores, float* __restrict__ partial,
                  float* __restrict__ out_zero, int N, int out_elems)
{
    __shared__ float red[256];
    const int tid = threadIdx.x;
    const long stride = (long)gridDim.x * 256;
    for (long o = (long)blockIdx.x * 256 + tid; o < out_elems; o += stride)
        out_zero[o] = 0.0f;
    float m = -3.402823466e38f;
    for (long i = (long)blockIdx.x * 256 + tid; i < N; i += stride)
        m = fmaxf(m, scores[i]);
    red[tid] = m;
    __syncthreads();
    for (int s = 128; s > 0; s >>= 1) {
        if (tid < s) red[tid] = fmaxf(red[tid], red[tid + s]);
        __syncthreads();
    }
    if (tid == 0) partial[blockIdx.x] = red[0];
}

// --------------- Pass C: final max + init seg sums and Z --------------------
__global__ void __launch_bounds__(1024)
finalize_max_kernel(const float* __restrict__ partial, int nPartial,
                    float* __restrict__ M, float* __restrict__ Z,
                    float* __restrict__ seg, int B)
{
    __shared__ float red[1024];
    const int tid = threadIdx.x;
    float m = -3.402823466e38f;
    for (int i = tid; i < nPartial; i += 1024) m = fmaxf(m, partial[i]);
    red[tid] = m;
    __syncthreads();
    for (int s = 512; s > 0; s >>= 1) {
        if (tid < s) red[tid] = fmaxf(red[tid], red[tid + s]);
        __syncthreads();
    }
    if (tid == 0) { M[0] = red[0]; Z[0] = 0.0f; }
    for (int i = tid; i < B; i += 1024) seg[i] = 0.0f;
}

// --------------- Pass D: Z = sum exp, seg[b] = segment exp sums -------------
__global__ void __launch_bounds__(256)
expsum_kernel(const float* __restrict__ scores, const int* __restrict__ batch,
              const float* __restrict__ M, float* __restrict__ Z,
              float* __restrict__ seg, int N)
{
    __shared__ float red[256];
    const int tid = threadIdx.x;
    const float m = M[0];
    const long stride = (long)gridDim.x * 256;
    float lz = 0.0f;
    for (long i = (long)blockIdx.x * 256 + tid; i < N; i += stride) {
        float e = __expf(scores[i] - m);
        lz += e;
        atomicAdd(&seg[batch[i]], e);
    }
    red[tid] = lz;
    __syncthreads();
    for (int s = 128; s > 0; s >>= 1) {
        if (tid < s) red[tid] += red[tid + s];
        __syncthreads();
    }
    if (tid == 0) atomicAdd(Z, red[0]);
}

// --------------- Pass D2: scores <- w_norm (in place) -----------------------
// w_norm = (e/Z) / (seg/Z + EPS) = e / (seg + EPS*Z)
__global__ void __launch_bounds__(256)
wnorm_kernel(float* __restrict__ scores, const int* __restrict__ batch,
             const float* __restrict__ M, const float* __restrict__ Z,
             const float* __restrict__ seg, int N)
{
    long i = (long)blockIdx.x * 256 + threadIdx.x;
    if (i < N) {
        float e = __expf(scores[i] - M[0]);
        scores[i] = e / (seg[batch[i]] + 1e-8f * Z[0]);
    }
}

// --------------- Pass E: out[b,:] += w_norm * x[i,:] ------------------------
__global__ void __launch_bounds__(256)
aggregate_kernel(const float* __restrict__ X, const float* __restrict__ wn,
                 const int* __restrict__ batch, float* __restrict__ out, int N)
{
    __shared__ float swn[CHUNK];
    __shared__ int   sb[CHUNK];
    const int tid = threadIdx.x;                 // column h = tid
    const long r0 = (long)blockIdx.x * CHUNK;
    long rem = (long)N - r0;
    const int rows = (rem < (long)CHUNK) ? (int)rem : CHUNK;
    for (int i = tid; i < rows; i += 256) { swn[i] = wn[r0 + i]; sb[i] = batch[r0 + i]; }
    __syncthreads();

    const float* xp = X + r0 * HDIM + tid;
    float acc = 0.0f;
    int curb = sb[0];
    int i = 0;
    // 4-row fast path when no segment boundary (batch is sorted)
    for (; i + 4 <= rows; ) {
        if (sb[i + 3] == curb) {
            float x0 = xp[(long)(i + 0) * HDIM];
            float x1 = xp[(long)(i + 1) * HDIM];
            float x2 = xp[(long)(i + 2) * HDIM];
            float x3 = xp[(long)(i + 3) * HDIM];
            acc += swn[i] * x0 + swn[i + 1] * x1 + swn[i + 2] * x2 + swn[i + 3] * x3;
            i += 4;
        } else {
            for (int j = 0; j < 4; ++j, ++i) {
                int b = sb[i];
                if (b != curb) { atomicAdd(&out[(long)curb * HDIM + tid], acc); acc = 0.0f; curb = b; }
                acc += swn[i] * xp[(long)i * HDIM];
            }
        }
    }
    for (; i < rows; ++i) {
        int b = sb[i];
        if (b != curb) { atomicAdd(&out[(long)curb * HDIM + tid], acc); acc = 0.0f; curb = b; }
        acc += swn[i] * xp[(long)i * HDIM];
    }
    atomicAdd(&out[(long)curb * HDIM + tid], acc);
}

// ---------------------------------------------------------------------------
extern "C" void kernel_launch(void* const* d_in, const int* in_sizes, int n_in,
                              void* d_out, int out_size, void* d_ws, size_t ws_size,
                              hipStream_t stream)
{
    const float* X   = (const float*)d_in[0];
    const float* W1  = (const float*)d_in[1];
    const float* b1  = (const float*)d_in[2];
    const float* W2  = (const float*)d_in[3];
    const float* b2  = (const float*)d_in[4];
    const int*   bat = (const int*)d_in[5];
    float* out = (float*)d_out;

    const int N = in_sizes[0] / HDIM;    // 1,000,000
    const int B = out_size / HDIM;       // 1024

    // Workspace layout (floats): scores[N] | partial[1024] | M | Z | seg[B]
    float* ws      = (float*)d_ws;
    float* scores  = ws;
    float* partial = ws + N;
    float* Mp      = partial + 1024;
    float* Zp      = Mp + 1;
    float* seg     = Zp + 1;

    const int numTiles = (N + 127) / 128;
    const int blocksA  = numTiles < 2048 ? numTiles : 2048;
    const size_t ldsA  = (size_t)HHALF * LDS_STRIDE * sizeof(__bf16);   // 67,584 B

    scores_kernel<<<blocksA, 256, ldsA, stream>>>(X, W1, b1, W2, b2, scores, N, numTiles);
    reduce_max_kernel<<<1024, 256, 0, stream>>>(scores, partial, out, N, out_size);
    finalize_max_kernel<<<1, 1024, 0, stream>>>(partial, 1024, Mp, Zp, seg, B);
    expsum_kernel<<<1024, 256, 0, stream>>>(scores, bat, Mp, Zp, seg, N);
    wnorm_kernel<<<(N + 255) / 256, 256, 0, stream>>>(scores, bat, Mp, Zp, seg, N);
    aggregate_kernel<<<(N + CHUNK - 1) / CHUNK, 256, 0, stream>>>(X, scores, bat, out, N);
    (void)n_in; (void)ws_size;
}
